// NTM_63806034149392
// MI455X (gfx1250) — compile-verified, matrix operand order checked by reading, not run on hardware
//
#include <hip/hip_runtime.h>
#include <hip/hip_bf16.h>
#include <math.h>

// ---------------- types for WMMA ----------------
typedef __attribute__((ext_vector_type(16))) __bf16 bf16x16;
typedef __attribute__((ext_vector_type(8)))  float  f32x8;
typedef __attribute__((ext_vector_type(4)))  float  f32x4;

union FragU { bf16x16 v; f32x4 q[2]; };

// scheduling-group hints (VMEM-read group ahead of WMMA group) if available
#if defined(__has_builtin)
#if __has_builtin(__builtin_amdgcn_sched_group_barrier)
#define SCHED_GROUP(mask, size, id) __builtin_amdgcn_sched_group_barrier(mask, size, id)
#endif
#endif
#ifndef SCHED_GROUP
#define SCHED_GROUP(mask, size, id)
#endif

// Load a 16x32 bf16 WMMA fragment (A, or B from [N,K] row-major weights).
// Lane L<16 : row = row0+L,   K = k0+{0..7, 16..23}
// Lane L>=16: row = row0+L-16, K = k0+{8..15, 24..31}
__device__ inline bf16x16 load_frag(const __bf16* p, int ld, int row0, int k0) {
    int lane = threadIdx.x & 31;
    const __bf16* q = p + (size_t)(row0 + (lane & 15)) * ld + k0 + ((lane >> 4) << 3);
    FragU u;
    u.q[0] = *(const f32x4*)q;          // 8 bf16 = 16B
    u.q[1] = *(const f32x4*)(q + 16);   // next 8 bf16, 32B later
    return u.v;
}

// C/D tile: VGPR i -> M = i (lanes 0-15) / 8+i (lanes 16-31); N = lane&15
__device__ inline void store_tile(float* out, int ld, int row0, int col0, const f32x8& d) {
    int lane = threadIdx.x & 31;
    int col  = col0 + (lane & 15);
    int rb   = row0 + ((lane >> 4) << 3);
#pragma unroll
    for (int i = 0; i < 8; ++i) out[(size_t)(rb + i) * ld + col] = d[i];
}

// fast transcendentals: v_exp_f32 path, no libm branch trees
__device__ inline float fsigm(float x)  { return 1.f / (1.f + __expf(-x)); }
__device__ inline float ftanh(float x)  { return 2.f / (1.f + __expf(-2.f * x)) - 1.f; }

// ---------------- weight conversion ----------------
__global__ void cvt_bf16(const float* __restrict__ s, __bf16* __restrict__ d, int n) {
    int i = blockIdx.x * blockDim.x + threadIdx.x;
    if (i < n) d[i] = (__bf16)s[i];
}

// Compact W_head: per head only cols 0..143 (129 used, padded to 9 WMMA tiles)
__global__ void cvt_head(const float* __restrict__ W, __bf16* __restrict__ d) {
    int i = blockIdx.x * blockDim.x + threadIdx.x;
    if (i >= 576 * 512) return;
    int r = i >> 9, k = i & 511;
    int head = r / 144, cc = r % 144;
    d[i] = (__bf16)W[(size_t)(head * 2115 + cc) * 512 + k];
}

// zero mem/h/c (live in d_out) and read carry
__global__ void init_zero(float* tail, float* readf) {
    int i = blockIdx.x * blockDim.x + threadIdx.x;
    if (i < 4227072) tail[i] = 0.f;
    if (i < 8192)    readf[i] = 0.f;
}

// ---------------- per-step kernels ----------------
// Build A_gates[32,1024] bf16 = [x_t(256) | read(256) | h(512)]
__global__ void prep_kernel(const float* __restrict__ x, const float* __restrict__ readf,
                            const float* __restrict__ h, __bf16* __restrict__ Ag, int t) {
    int i = blockIdx.x * blockDim.x + threadIdx.x;
    if (i >= 32 * 1024) return;
    int m = i >> 10, k = i & 1023;
    float v;
    if (k < 256)      v = x[(size_t)(m * 32 + t) * 256 + k];
    else if (k < 512) v = readf[m * 256 + (k - 256)];
    else              v = h[m * 512 + (k - 512)];
    Ag[i] = (__bf16)v;
}

// gates = Ag @ [W_ih;W_hh]^T + biases, fused LSTM cell. 64 waves, 4 gate tiles/wave.
// K loop double-buffered; sched-group hints keep next-iter loads ahead of WMMAs.
__global__ void gates_lstm_kernel(const __bf16* __restrict__ Ag,
                                  const __bf16* __restrict__ Wih, const __bf16* __restrict__ Whh,
                                  const float* __restrict__ bih, const float* __restrict__ bhh,
                                  float* __restrict__ h, float* __restrict__ c,
                                  __bf16* __restrict__ act) {
    int wg = blockIdx.x * 8 + (threadIdx.x >> 5);   // 0..63
    int lane = threadIdx.x & 31;
    int mt = wg >> 5;   // 0..1
    int nt = wg & 31;   // 0..31  (within H=512)
    f32x8 acc[4];
#pragma unroll
    for (int q = 0; q < 4; ++q) {
        int col = q * 512 + nt * 16 + (lane & 15);
        float bv = bih[col] + bhh[col];
#pragma unroll
        for (int i = 0; i < 8; ++i) acc[q][i] = bv;
    }
    // prologue: fragments for kk = 0
    bf16x16 a_cur = load_frag(Ag, 1024, mt * 16, 0);
    bf16x16 b_cur[4];
#pragma unroll
    for (int q = 0; q < 4; ++q) b_cur[q] = load_frag(Wih, 512, q * 512 + nt * 16, 0);
#pragma unroll
    for (int kk = 0; kk < 32; ++kk) {   // K=1024: 16 chunks W_ih + 16 chunks W_hh
        bf16x16 a_nxt = a_cur;
        bf16x16 b_nxt[4];
#pragma unroll
        for (int q = 0; q < 4; ++q) b_nxt[q] = b_cur[q];
        int kn = kk + 1;
        if (kn < 32) {
            a_nxt = load_frag(Ag, 1024, mt * 16, kn * 32);
            const __bf16* W = (kn < 16) ? Wih : Whh;
            int bk = (kn < 16) ? kn * 32 : (kn - 16) * 32;
#pragma unroll
            for (int q = 0; q < 4; ++q) b_nxt[q] = load_frag(W, 512, q * 512 + nt * 16, bk);
        }
#pragma unroll
        for (int q = 0; q < 4; ++q)
            acc[q] = __builtin_amdgcn_wmma_f32_16x16x32_bf16(
                false, a_cur, false, b_cur[q], (short)0, acc[q], false, false);
        a_cur = a_nxt;
#pragma unroll
        for (int q = 0; q < 4; ++q) b_cur[q] = b_nxt[q];
        // per-iteration schedule: 10 VMEM reads (next iter frags), then 4 WMMAs
        SCHED_GROUP(0x020, 10, 0);
        SCHED_GROUP(0x008, 4, 0);
    }
    // fused LSTM elementwise: i=acc0 f=acc1 g=acc2 o=acc3
    int col = nt * 16 + (lane & 15);
    int rb  = mt * 16 + ((lane >> 4) << 3);
#pragma unroll
    for (int i = 0; i < 8; ++i) {
        int m = rb + i;
        float iv = fsigm(acc[0][i]);
        float fv = fsigm(acc[1][i]);
        float gv = ftanh(acc[2][i]);
        float ov = fsigm(acc[3][i]);
        float cn = fv * c[m * 512 + col] + iv * gv;
        float hn = ov * ftanh(cn);
        c[m * 512 + col] = cn;
        h[m * 512 + col] = hn;
        act[m * 768 + col] = (__bf16)hn;    // bf16 h for head/out GEMMs
    }
}

// hp[32, 4*144] = h @ Whead_compact^T + b_head (only 144 cols/head). 72 waves.
__global__ void head_kernel(const __bf16* __restrict__ act, const __bf16* __restrict__ Whd,
                            const float* __restrict__ bhd, float* __restrict__ hp) {
    int wg = blockIdx.x * 8 + (threadIdx.x >> 5);   // 0..71
    int lane = threadIdx.x & 31;
    int mt = wg % 2;
    int tile = wg / 2;        // 0..35
    int head = tile / 9;
    int col0 = (tile % 9) * 16;
    f32x8 acc;
    float bv = bhd[head * 2115 + col0 + (lane & 15)];
#pragma unroll
    for (int i = 0; i < 8; ++i) acc[i] = bv;
    bf16x16 a_cur = load_frag(act, 768, mt * 16, 0);
    bf16x16 b_cur = load_frag(Whd, 512, head * 144 + col0, 0);
#pragma unroll
    for (int kk = 0; kk < 16; ++kk) {
        bf16x16 a_nxt = a_cur, b_nxt = b_cur;
        if (kk + 1 < 16) {
            a_nxt = load_frag(act, 768, mt * 16, (kk + 1) * 32);
            b_nxt = load_frag(Whd, 512, head * 144 + col0, (kk + 1) * 32);
        }
        acc = __builtin_amdgcn_wmma_f32_16x16x32_bf16(
            false, a_cur, false, b_cur, (short)0, acc, false, false);
        a_cur = a_nxt; b_cur = b_nxt;
        SCHED_GROUP(0x020, 4, 0);
        SCHED_GROUP(0x008, 1, 0);
    }
    store_tile(hp, 576, mt * 16, head * 144 + col0, acc);
}

// NTM attention: one block per batch b (owns mem[b] exclusively). 512 threads.
#define MSZ 2048
__global__ void attn_kernel(const float* __restrict__ hp, float* __restrict__ mem,
                            float* __restrict__ scores, float* __restrict__ readf,
                            __bf16* __restrict__ act) {
    int b = blockIdx.x;
    int tid = threadIdx.x;
    __shared__ float keys[8][64];   // 0..3 read keys, 4..7 write keys
    __shared__ float swk[4][64];    // strength * write_key
    __shared__ float inv[8];
    __shared__ float red[4][8][64];
    const float* hpb = hp + b * 576;
    {
        int kk = tid >> 6, hh = tid & 63;
        int n = kk & 3;
        float v = (kk < 4) ? hpb[n * 144 + hh] : hpb[n * 144 + 64 + hh];
        keys[kk][hh] = v;
        if (kk >= 4) swk[n][hh] = fsigm(hpb[n * 144 + 128]) * v;
    }
    __syncthreads();
    // Phase A: scores[8][2048] = keys . mem rows, scaled
    float* scb = scores + (size_t)b * 8 * MSZ;
    float* memb = mem + (size_t)b * MSZ * 64;
    for (int m = tid; m < MSZ; m += 512) {
        const f32x4* mp = (const f32x4*)(memb + (size_t)m * 64);
        if (m + 512 < MSZ) __builtin_prefetch(memb + (size_t)(m + 512) * 64, 0, 1);
        f32x4 mv[16];
#pragma unroll
        for (int j = 0; j < 16; ++j) mv[j] = mp[j];
#pragma unroll
        for (int kk = 0; kk < 8; ++kk) {
            float acc = 0.f;
#pragma unroll
            for (int j = 0; j < 16; ++j)
                acc += mv[j][0] * keys[kk][j * 4 + 0] + mv[j][1] * keys[kk][j * 4 + 1]
                     + mv[j][2] * keys[kk][j * 4 + 2] + mv[j][3] * keys[kk][j * 4 + 3];
            scb[kk * MSZ + m] = acc * 0.125f;   // 1/sqrt(64)
        }
    }
    __syncthreads();
    // Phase B: softmax per score row; waves 0..7, one key each (wave32)
    int wv = tid >> 5, lane = tid & 31;
    if (wv < 8) {
        float* row = scb + wv * MSZ;
        float mx = -3.4e38f;
        for (int m = lane; m < MSZ; m += 32) mx = fmaxf(mx, row[m]);
#pragma unroll
        for (int off = 16; off > 0; off >>= 1) mx = fmaxf(mx, __shfl_xor(mx, off, 32));
        float sum = 0.f;
        for (int m = lane; m < MSZ; m += 32) {
            float e = __expf(row[m] - mx);    // single v_exp_f32
            row[m] = e;
            sum += e;
        }
#pragma unroll
        for (int off = 16; off > 0; off >>= 1) sum += __shfl_xor(sum, off, 32);
        if (lane == 0) inv[wv] = 1.f / sum;
    }
    __syncthreads();
    // Phase C: read = p_r @ mem (old), mem += p_w^T @ (s*wk). thread = (h, m-group)
    int hh = tid & 63, g = tid >> 6;
    float iR[4] = {inv[0], inv[1], inv[2], inv[3]};
    float iW[4] = {inv[4], inv[5], inv[6], inv[7]};
    float sw[4] = {swk[0][hh], swk[1][hh], swk[2][hh], swk[3][hh]};
    float accR[4] = {0.f, 0.f, 0.f, 0.f};
    for (int m = g; m < MSZ; m += 8) {
        float v = memb[(size_t)m * 64 + hh];
        float delta = 0.f;
#pragma unroll
        for (int n = 0; n < 4; ++n) {
            accR[n] += scb[n * MSZ + m] * iR[n] * v;
            delta   += scb[(4 + n) * MSZ + m] * iW[n] * sw[n];
        }
        memb[(size_t)m * 64 + hh] = v + delta;
    }
#pragma unroll
    for (int n = 0; n < 4; ++n) red[n][g][hh] = accR[n];
    __syncthreads();
    if (tid < 256) {
        int n = tid >> 6, h2 = tid & 63;
        float s = 0.f;
#pragma unroll
        for (int gg = 0; gg < 8; ++gg) s += red[n][gg][h2];
        readf[b * 256 + n * 64 + h2] = s;
        act[b * 768 + 512 + n * 64 + h2] = (__bf16)s;   // bf16 read for out GEMM
    }
}

// out[b, t, :] = [h|read] @ W_out^T + b_out. 32 waves.
__global__ void out_gemm_kernel(const __bf16* __restrict__ act, const __bf16* __restrict__ Wout,
                                const float* __restrict__ bout, float* __restrict__ outputs, int t) {
    int wg = blockIdx.x * 8 + (threadIdx.x >> 5);   // 0..31
    int lane = threadIdx.x & 31;
    int mt = wg & 1;
    int nt = wg >> 1;   // 0..15
    f32x8 acc;
    float bv = bout[nt * 16 + (lane & 15)];
#pragma unroll
    for (int i = 0; i < 8; ++i) acc[i] = bv;
    bf16x16 a_cur = load_frag(act, 768, mt * 16, 0);
    bf16x16 b_cur = load_frag(Wout, 768, nt * 16, 0);
#pragma unroll
    for (int kk = 0; kk < 24; ++kk) {   // K = 768
        bf16x16 a_nxt = a_cur, b_nxt = b_cur;
        if (kk + 1 < 24) {
            a_nxt = load_frag(act, 768, mt * 16, (kk + 1) * 32);
            b_nxt = load_frag(Wout, 768, nt * 16, (kk + 1) * 32);
        }
        acc = __builtin_amdgcn_wmma_f32_16x16x32_bf16(
            false, a_cur, false, b_cur, (short)0, acc, false, false);
        a_cur = a_nxt; b_cur = b_nxt;
        SCHED_GROUP(0x020, 4, 0);
        SCHED_GROUP(0x008, 1, 0);
    }
    int col = nt * 16 + (lane & 15);
    int rb  = mt * 16 + ((lane >> 4) << 3);
#pragma unroll
    for (int i = 0; i < 8; ++i)
        outputs[(size_t)(rb + i) * 8192 + t * 256 + col] = acc[i];
}

// ---------------- launcher ----------------
extern "C" void kernel_launch(void* const* d_in, const int* in_sizes, int n_in,
                              void* d_out, int out_size, void* d_ws, size_t ws_size,
                              hipStream_t stream) {
    (void)in_sizes; (void)n_in; (void)out_size; (void)ws_size;
    const float* x      = (const float*)d_in[0];   // [32,32,256]
    const float* W_ih   = (const float*)d_in[1];   // [2048,512]
    const float* W_hh   = (const float*)d_in[2];   // [2048,512]
    const float* b_ih   = (const float*)d_in[3];
    const float* b_hh   = (const float*)d_in[4];
    const float* W_head = (const float*)d_in[5];   // [8460,512]
    const float* b_head = (const float*)d_in[6];
    const float* W_out  = (const float*)d_in[7];   // [256,768]
    const float* b_out  = (const float*)d_in[8];

    // d_out = outputs[32,32,256] | mem[32,2048,64] | h[32,512] | c[32,512]
    float* outputs = (float*)d_out;
    float* mem     = outputs + 262144;
    float* hbuf    = mem + 4194304;
    float* cbuf    = hbuf + 16384;

    // workspace carve (256B aligned)
    uintptr_t w = (uintptr_t)d_ws;
    auto take = [&](size_t bytes) { uintptr_t p = w; w += (bytes + 255) & ~(size_t)255; return (void*)p; };
    float*  readf  = (float*) take(8192 * 4);          // read carry [32,4,64]
    float*  hp     = (float*) take(18432 * 4);         // [32, 4*144]
    float*  scores = (float*) take((size_t)32 * 8 * MSZ * 4);
    __bf16* Ag     = (__bf16*)take(32768 * 2);         // [32,1024]
    __bf16* act    = (__bf16*)take(24576 * 2);         // [32,768] = [h|read] bf16
    __bf16* Wih_b  = (__bf16*)take((size_t)1048576 * 2);
    __bf16* Whh_b  = (__bf16*)take((size_t)1048576 * 2);
    __bf16* Whd_b  = (__bf16*)take((size_t)294912 * 2);
    __bf16* Wout_b = (__bf16*)take((size_t)196608 * 2);

    // one-time (idempotent) weight conversion + state zeroing
    cvt_bf16<<<(1048576 + 255) / 256, 256, 0, stream>>>(W_ih, Wih_b, 1048576);
    cvt_bf16<<<(1048576 + 255) / 256, 256, 0, stream>>>(W_hh, Whh_b, 1048576);
    cvt_head<<<(294912 + 255) / 256, 256, 0, stream>>>(W_head, Whd_b);
    cvt_bf16<<<(196608 + 255) / 256, 256, 0, stream>>>(W_out, Wout_b, 196608);
    init_zero<<<(4227072 + 255) / 256, 256, 0, stream>>>(mem, readf);

    for (int t = 0; t < 32; ++t) {
        prep_kernel<<<128, 256, 0, stream>>>(x, readf, hbuf, Ag, t);
        gates_lstm_kernel<<<8, 256, 0, stream>>>(Ag, Wih_b, Whh_b, b_ih, b_hh, hbuf, cbuf, act);
        head_kernel<<<9, 256, 0, stream>>>(act, Whd_b, b_head, hp);
        attn_kernel<<<32, 512, 0, stream>>>(hp, mem, scores, readf, act);
        out_gemm_kernel<<<4, 256, 0, stream>>>(act, Wout_b, b_out, outputs, t);
    }
}
